// Qwen3Attention_73796128080147
// MI455X (gfx1250) — matrix operand order repaired and checked by reading, not compiled, not run on hardware
//
#include <hip/hip_runtime.h>
#include <cstdint>

// ---------------------------------------------------------------- types
typedef __attribute__((ext_vector_type(16))) __bf16 v16bf;
typedef __attribute__((ext_vector_type(8)))  float  v8f;
typedef unsigned short u16;

union U16x16 { v16bf v; uint4 u[2]; };   // 32 bytes: one WMMA bf16 A/B operand per lane

#define T_TOK   4096
#define HID     2048
#define QKV_OUT 4096   // 2048 q + 1024 k + 1024 v
#define HD      128
#define NH      16
#define NKV     8

__device__ inline u16 f2bf(float f) {
  unsigned int u = __float_as_uint(f);
  u += 0x7FFFu + ((u >> 16) & 1u);          // round-to-nearest-even
  return (u16)(u >> 16);
}

__device__ inline v8f wmma_bf16(v16bf a, v16bf b, v8f c) {
  // D = A(16x32 bf16) * B(32x16 bf16) + C(16x16 f32)
  return __builtin_amdgcn_wmma_f32_16x16x32_bf16(false, a, false, b, (short)0, c, false, false);
}

__device__ inline v8f zero8() {
  v8f z;
#pragma unroll
  for (int i = 0; i < 8; ++i) z[i] = 0.0f;
  return z;
}

__device__ inline float red16_max(float v) {
#pragma unroll
  for (int m = 1; m <= 8; m <<= 1) v = fmaxf(v, __shfl_xor(v, m, 32));
  return v; // reduced within each 16-lane half of the wave32
}
__device__ inline float red16_sum(float v) {
#pragma unroll
  for (int m = 1; m <= 8; m <<= 1) v += __shfl_xor(v, m, 32);
  return v;
}

// ---------------------------------------------------------------- fp32 -> bf16
__global__ void cvt_bf16(const float* __restrict__ s, u16* __restrict__ d, int n) {
  int i = blockIdx.x * blockDim.x + threadIdx.x;
  int stride = gridDim.x * blockDim.x;
  for (; i < n; i += stride) d[i] = f2bf(s[i]);
}

// ---------------------------------------------------------------- bf16 GEMM: C[M,N] = A[M,K] * B[N,K]^T
// One wave owns a 32(M) x 64(N) tile: per K-step of 32, two A operands share four B
// operands -> 8 WMMAs per 6KB of L2 traffic (~21 FLOP/B).
__global__ __launch_bounds__(256) void gemm_bf16(const u16* __restrict__ A,
                                                 const u16* __restrict__ B,
                                                 float* __restrict__ C,
                                                 int M, int N, int K) {
  const int lane = threadIdx.x & 31;
  const int wid  = blockIdx.x * (blockDim.x >> 5) + (threadIdx.x >> 5);
  const int ntiles = N >> 6;
  const int mt = wid / ntiles, nt = wid % ntiles;
  const int mBase = mt << 5, nBase = nt << 6;
  if (mBase >= M) return;

  const int m  = lane & 15;
  const int hi = lane >> 4;                     // 0: lanes 0-15, 1: lanes 16-31
  const u16* Arow0 = A + (size_t)(mBase + m) * K;
  const u16* Arow1 = A + (size_t)(mBase + 16 + m) * K;

  v8f acc[8];                                   // [mi*4 + sub]
#pragma unroll
  for (int i = 0; i < 8; ++i) acc[i] = zero8();

  for (int k0 = 0; k0 < K; k0 += 32) {
    // A operands: lane<16 holds row m, K = k0+{0..7,16..23}; lane>=16 holds K = k0+{8..15,24..31}
    const int bk = k0 + hi * 8;
    U16x16 a0, a1;
    a0.u[0] = *(const uint4*)(Arow0 + bk);
    a0.u[1] = *(const uint4*)(Arow0 + bk + 16);
    a1.u[0] = *(const uint4*)(Arow1 + bk);
    a1.u[1] = *(const uint4*)(Arow1 + bk + 16);

    if (k0 + 32 < K) {                          // uniform; next K-step prefetch
      __builtin_prefetch(Arow0 + bk + 32, 0, 3);
      __builtin_prefetch(Arow1 + bk + 32, 0, 3);
    }

#pragma unroll
    for (int sub = 0; sub < 4; ++sub) {
      // B operand: lane n holds column (nBase+sub*16+m), K = k0 + hi*8 + {0..7,16..23} (matches A)
      const u16* Brow = B + (size_t)(nBase + sub * 16 + m) * K + k0 + hi * 8;
      U16x16 b;
      b.u[0] = *(const uint4*)(Brow);
      b.u[1] = *(const uint4*)(Brow + 16);
      acc[sub]     = wmma_bf16(a0.v, b.v, acc[sub]);
      acc[4 + sub] = wmma_bf16(a1.v, b.v, acc[4 + sub]);
    }
  }

#pragma unroll
  for (int mi = 0; mi < 2; ++mi)
#pragma unroll
    for (int sub = 0; sub < 4; ++sub)
#pragma unroll
      for (int r = 0; r < 8; ++r) {
        const int row = mBase + mi * 16 + r + hi * 8;
        const int col = nBase + sub * 16 + m;
        C[(size_t)row * N + col] = acc[mi * 4 + sub][r];
      }
}

// ---------------------------------------------------------------- RMS-norm + RoPE + layout shuffle
// grid = T * 32 blocks of 128 threads. head<16: q, head<24: k, else: v -> vT[kvh][d][t]
__global__ __launch_bounds__(128) void normrope(const float* __restrict__ qkv,
                                                const float* __restrict__ qnw,
                                                const float* __restrict__ knw,
                                                const int*   __restrict__ pos,
                                                u16* __restrict__ Qo,   // [T, 2048]
                                                u16* __restrict__ Ko,   // [T, 1024]
                                                u16* __restrict__ VTo)  // [1024, T]
{
  const int t    = blockIdx.x >> 5;
  const int head = blockIdx.x & 31;
  const int d    = threadIdx.x;          // 0..127
  __shared__ float red[128];

  if (head < 24) {
    const bool isQ = head < 16;
    const int  h   = isQ ? head : head - 16;
    const int  colBase = isQ ? h * HD : HID + h * HD;
    float v = qkv[(size_t)t * QKV_OUT + colBase + d];

    red[d] = v * v;
    __syncthreads();
#pragma unroll
    for (int s = 64; s > 0; s >>= 1) {
      if (d < s) red[d] += red[d + s];
      __syncthreads();
    }
    const float rms = rsqrtf(red[0] * (1.0f / HD) + 1e-6f);
    __syncthreads();

    const float w = isQ ? qnw[d] : knw[d];
    const float nv = v * rms * w;
    red[d] = nv;
    __syncthreads();
    const float other = red[d ^ 64];

    const int   i   = d & 63;
    const float inv = __powf(1.0e6f, -(float)i * (1.0f / 64.0f));
    const float ang = (float)pos[t] * inv;
    float sn, cs;
    __sincosf(ang, &sn, &cs);
    const float out = (d < 64) ? nv * cs - other * sn : nv * cs + other * sn;

    if (isQ) Qo[(size_t)t * (NH * HD) + h * HD + d] = f2bf(out);
    else     Ko[(size_t)t * (NKV * HD) + h * HD + d] = f2bf(out);
  } else {
    const int kh = head - 24;
    const float v = qkv[(size_t)t * QKV_OUT + HID + NKV * HD + kh * HD + d];
    VTo[(size_t)(kh * HD + d) * T_TOK + t] = f2bf(v);   // transposed for PV B-operand
  }
}

// ---------------------------------------------------------------- flash attention
// 1 wave per (head, 16-row q tile). s-steps of 32: 8 QK wmma + online softmax + 8 PV wmma.
__global__ __launch_bounds__(128) void flash(const u16* __restrict__ Q,   // [T, 2048]
                                             const u16* __restrict__ Kb,  // [T, 1024]
                                             const u16* __restrict__ Vt,  // [1024, T]
                                             u16* __restrict__ O)         // [T, 2048]
{
  __shared__ __align__(16) u16 plds[4][16 * 32];   // 1KB P-tile per wave
  const int lane  = threadIdx.x & 31;
  const int wslot = threadIdx.x >> 5;
  const int wid   = blockIdx.x * 4 + wslot;        // 0..4095
  const int qt = wid & 255;
  const int h  = wid >> 8;
  const int kh = h >> 1;                            // GQA: 2 q heads per kv head
  const int qbase = qt << 4;
  const int smax  = qbase + 15;
  const float scale = 0.088388347648318447f;        // 1/sqrt(128)

  const int m  = lane & 15;
  const int hi = lane >> 4;

  // Q A-operands for the 4 K-steps over d (resident for whole kernel)
  U16x16 aq[4];
  {
    const u16* qrow = Q + (size_t)(qbase + m) * (NH * HD) + h * HD;
#pragma unroll
    for (int kk = 0; kk < 4; ++kk) {
      const int bd = kk * 32 + hi * 8;
      aq[kk].u[0] = *(const uint4*)(qrow + bd);
      aq[kk].u[1] = *(const uint4*)(qrow + bd + 16);
    }
  }

  v8f acc[8];
#pragma unroll
  for (int dt = 0; dt < 8; ++dt) acc[dt] = zero8();
  float mrow[8], lrow[8];
#pragma unroll
  for (int r = 0; r < 8; ++r) { mrow[r] = -3.0e38f; lrow[r] = 0.0f; }

  for (int sb = 0; sb <= smax; sb += 32) {
    // ---- scores: two 16x16 tiles (cols sb..sb+15, sb+16..sb+31)
    v8f S[2] = {zero8(), zero8()};
#pragma unroll
    for (int c = 0; c < 2; ++c) {
      const int colBase = sb + c * 16;
      if (colBase <= smax) {                         // wave-uniform; avoids OOB rows
        const u16* kr = Kb + (size_t)(colBase + m) * (NKV * HD) + kh * HD + hi * 8;
#pragma unroll
        for (int kk = 0; kk < 4; ++kk) {
          U16x16 b;
          b.u[0] = *(const uint4*)(kr + kk * 32);
          b.u[1] = *(const uint4*)(kr + kk * 32 + 16);
          S[c] = wmma_bf16(aq[kk].v, b.v, S[c]);
        }
      }
    }

    // ---- causal mask + online softmax (row M = r + hi*8, col N = lane&15)
#pragma unroll
    for (int r = 0; r < 8; ++r) {
      const int trow = qbase + r + hi * 8;
      const int s0c = sb + m, s1c = sb + 16 + m;
      float v0 = (s0c <= trow) ? S[0][r] * scale : -1.0e9f;
      float v1 = (s1c <= trow) ? S[1][r] * scale : -1.0e9f;
      const float rm   = red16_max(fmaxf(v0, v1));
      const float mnew = fmaxf(mrow[r], rm);
      const float alpha = __expf(mrow[r] - mnew);
      const float p0 = __expf(v0 - mnew);
      const float p1 = __expf(v1 - mnew);
      lrow[r] = lrow[r] * alpha + red16_sum(p0 + p1);
      mrow[r] = mnew;
#pragma unroll
      for (int dt = 0; dt < 8; ++dt) acc[dt][r] *= alpha;
      plds[wslot][(r + hi * 8) * 32 + m]      = f2bf(p0);
      plds[wslot][(r + hi * 8) * 32 + 16 + m] = f2bf(p1);
    }

    asm volatile("s_wait_dscnt 0" ::: "memory");     // same-wave LDS store->load ordering

    // ---- P as A-operand (16x32 bf16) from LDS
    U16x16 pa;
    {
      const u16* lp = &plds[wslot][m * 32 + hi * 8];
      pa.u[0] = *(const uint4*)(lp);
      pa.u[1] = *(const uint4*)(lp + 16);
    }

    // ---- PV: 8 d-tiles of 16, B from vT (K = s contiguous)
#pragma unroll
    for (int dt = 0; dt < 8; ++dt) {
      const u16* vr = Vt + (size_t)(kh * HD + dt * 16 + m) * T_TOK + sb + hi * 8;
      U16x16 b;
      b.u[0] = *(const uint4*)(vr);
      b.u[1] = *(const uint4*)(vr + 16);
      acc[dt] = wmma_bf16(pa.v, b.v, acc[dt]);
    }
  }

  // ---- epilogue: normalize and store o as bf16 [T, 2048]
#pragma unroll
  for (int dt = 0; dt < 8; ++dt)
#pragma unroll
    for (int r = 0; r < 8; ++r) {
      const int row = qbase + r + hi * 8;
      const int col = h * HD + dt * 16 + m;
      O[(size_t)row * (NH * HD) + col] = f2bf(acc[dt][r] / lrow[r]);
    }
}

// ---------------------------------------------------------------- host launcher
extern "C" void kernel_launch(void* const* d_in, const int* in_sizes, int n_in,
                              void* d_out, int out_size, void* d_ws, size_t ws_size,
                              hipStream_t stream) {
  const float* x     = (const float*)d_in[0];   // [4096, 2048]
  const float* qkv_w = (const float*)d_in[1];   // [4096, 2048]
  const float* qnw   = (const float*)d_in[2];   // [128]
  const float* knw   = (const float*)d_in[3];   // [128]
  const float* o_w   = (const float*)d_in[4];   // [2048, 2048]
  const int*   pos   = (const int*)d_in[5];     // [4096]

  char* ws = (char*)d_ws;
  u16*   xb   = (u16*)ws;            ws += (size_t)T_TOK * HID * 2;       // 16 MB
  u16*   wb   = (u16*)ws;            ws += (size_t)QKV_OUT * HID * 2;     // 16 MB
  u16*   owb  = (u16*)ws;            ws += (size_t)HID * HID * 2;         //  8 MB
  float* qkvf = (float*)ws;          ws += (size_t)T_TOK * QKV_OUT * 4;   // 64 MB
  u16*   qb   = (u16*)ws;            ws += (size_t)T_TOK * NH * HD * 2;   // 16 MB
  u16*   kb   = (u16*)ws;            ws += (size_t)T_TOK * NKV * HD * 2;  //  8 MB
  u16*   vtb  = (u16*)ws;            ws += (size_t)NKV * HD * T_TOK * 2;  //  8 MB
  u16*   ob   = (u16*)ws;            ws += (size_t)T_TOK * NH * HD * 2;   // 16 MB

  // 1) fp32 -> bf16 operand conversion
  cvt_bf16<<<4096, 256, 0, stream>>>(x,     xb,  T_TOK * HID);
  cvt_bf16<<<4096, 256, 0, stream>>>(qkv_w, wb,  QKV_OUT * HID);
  cvt_bf16<<<2048, 256, 0, stream>>>(o_w,   owb, HID * HID);

  // 2) QKV projection: qkvf[4096,4096] = x * qkv_w^T   (32x64 tiles: 128*64 waves, 8/block)
  gemm_bf16<<<1024, 256, 0, stream>>>(xb, wb, qkvf, T_TOK, QKV_OUT, HID);

  // 3) RMS-norm + RoPE + V transpose
  normrope<<<T_TOK * 32, 128, 0, stream>>>(qkvf, qnw, knw, pos, qb, kb, vtb);

  // 4) causal flash attention (4096 waves, 4 per block)
  flash<<<1024, 128, 0, stream>>>(qb, kb, vtb, ob);

  // 5) output projection: out[4096,2048] = o * o_w^T   (128*32 waves, 8/block)
  gemm_bf16<<<512, 256, 0, stream>>>(ob, owb, (float*)d_out, T_TOK, HID, HID);
}
